// AttnAligner_62096637166408
// MI455X (gfx1250) — matrix operand order repaired and checked by reading, not compiled, not run on hardware
//
#include <hip/hip_runtime.h>

// ---------------------------------------------------------------------------
// Types for CDNA5 WMMA
// ---------------------------------------------------------------------------
typedef __attribute__((ext_vector_type(16))) __bf16       v16bf;
typedef __attribute__((ext_vector_type(8)))  float        v8f;
typedef __attribute__((ext_vector_type(4)))  unsigned int v4u;
typedef __attribute__((ext_vector_type(4)))  int          v4i;

// Async global->LDS (CDNA5). Guarded so a missing builtin falls back to a
// synchronous load+store and the build still succeeds.
#if defined(__HIP_DEVICE_COMPILE__) && defined(__has_builtin)
#if __has_builtin(__builtin_amdgcn_global_load_async_to_lds_b128)
#define HAVE_ASYNC_LDS 1
#endif
#endif
#ifndef HAVE_ASYNC_LDS
#define HAVE_ASYNC_LDS 0
#endif

#if HAVE_ASYNC_LDS
typedef v4i __attribute__((address_space(1))) gas_v4i;   // global
typedef v4i __attribute__((address_space(3))) las_v4i;   // LDS
#endif

__device__ __forceinline__ void copy16_g2l(const __bf16* g, __bf16* l) {
#if HAVE_ASYNC_LDS
  __builtin_amdgcn_global_load_async_to_lds_b128(
      (gas_v4i*)g, (las_v4i*)l, 0, 0);
#else
  *(v4u*)l = *(const v4u*)g;
#endif
}

__device__ __forceinline__ void async_wait_all() {
#if HAVE_ASYNC_LDS
#if __has_builtin(__builtin_amdgcn_s_wait_asynccnt)
  __builtin_amdgcn_s_wait_asynccnt(0);
#else
  asm volatile("s_wait_asynccnt 0x0" ::: "memory");
#endif
#endif
}

__device__ __forceinline__ __bf16 f2bf(float f) {
  unsigned u = __builtin_bit_cast(unsigned, f);
  u += 0x7FFFu + ((u >> 16) & 1u);            // round to nearest even
  unsigned short s = (unsigned short)(u >> 16);
  return __builtin_bit_cast(__bf16, s);
}

__device__ __forceinline__ v16bf mkfrag(v4u lo, v4u hi) {
  union { struct { v4u a, b; } p; v16bf v; } u;
  u.p.a = lo; u.p.b = hi; return u.v;
}

// ---------------------------------------------------------------------------
// f32 -> bf16 conversion (one-time for weights)
// ---------------------------------------------------------------------------
__global__ __launch_bounds__(256)
void cvt_kernel(const float* __restrict__ s, __bf16* __restrict__ d, long long n) {
  const long long stride = (long long)gridDim.x * 256;
  for (long long i = (long long)blockIdx.x * 256 + threadIdx.x; i < n; i += stride)
    d[i] = f2bf(s[i]);
}

// ---------------------------------------------------------------------------
// Embedding via gather (scatter-add collapsed) + positional encoding
// ---------------------------------------------------------------------------
__global__ __launch_bounds__(256)
void embed_kernel(const float* __restrict__ attn, const int* __restrict__ tgt,
                  const float* __restrict__ emb, float* __restrict__ x,
                  __bf16* __restrict__ xbf) {
  __shared__ int   tok[64];
  __shared__ float aw[64];
  const int bl = blockIdx.x;
  const int b = bl >> 8, l = bl & 255;
  const int tid = threadIdx.x;
  if (tid < 64) {
    tok[tid] = tgt[b * 64 + tid];
    aw[tid]  = attn[((long long)(b * 64 + tid)) * 256 + l];
  }
  __syncthreads();
  const int d0 = tid * 2;
  float a0 = 0.f, a1 = 0.f;
  for (int tt = 0; tt < 64; ++tt) {
    const int tk = tok[tt];
    if (tk != 0) {
      const float2 w = *(const float2*)&emb[(long long)tk * 512 + d0];
      a0 += aw[tt] * w.x;
      a1 += aw[tt] * w.y;
    }
  }
  const float rate = __expf((float)d0 * (-9.210340371976184f / 512.f)); // ln(1e4)
  const float ang  = (float)l * rate;
  const float sc   = 22.62741699796952f;                                // sqrt(512)
  const float v0 = a0 * sc + __sinf(ang);
  const float v1 = a1 * sc + __cosf(ang);
  const long long o = (long long)bl * 512 + d0;
  x[o] = v0;  x[o + 1] = v1;
  xbf[o] = f2bf(v0); xbf[o + 1] = f2bf(v1);
}

// ---------------------------------------------------------------------------
// Strided/batched bf16 WMMA GEMM, 128x64 block tile, 32x32 per wave,
// double-buffered LDS with async global->LDS staging.
//   BPATH 0: B has sBk==1 (k contiguous)   -> direct b128 async stage
//   BPATH 1: B has sBn==1 (n contiguous)   -> vector load + b16 scatter
// Requires: M%128==0, N%64==0, K%32==0, element offsets 8-aligned, sAk==1.
// ---------------------------------------------------------------------------
template <int BPATH, bool RELU>
__global__ __launch_bounds__(256)
void gemm_bf16_kernel(const __bf16* __restrict__ A, const __bf16* __restrict__ B,
                      const float* __restrict__ bias,
                      float* __restrict__ Cf, __bf16* __restrict__ Cbf,
                      int M, int N, int K,
                      int sAm, long long bsAo, long long bsAi,
                      int sBk, int sBn, long long bsBo, long long bsBi,
                      int ldC, long long bsCo, long long bsCi,
                      int innerB) {
  __shared__ __bf16 As[2][128][40];   // +8 pad keeps rows 16B aligned
  __shared__ __bf16 Bs[2][64][40];

  const int t  = threadIdx.x;
  const int n0 = blockIdx.x * 64;
  const int m0 = blockIdx.y * 128;
  const int bb = blockIdx.z;
  const int outer = bb / innerB, inner = bb - outer * innerB;
  const long long offA = outer * bsAo + inner * bsAi;
  const long long offB = outer * bsBo + inner * bsBi;
  const long long offC = outer * bsCo + inner * bsCi;

  const int lane   = t & 31;
  const int wave   = t >> 5;
  const int wmBase = (wave & 3) * 32;   // 4 wave-rows  * 32 = 128
  const int wnBase = (wave >> 2) * 32;  // 2 wave-cols  * 32 = 64
  const int l15    = lane & 15;
  const int lh     = lane >> 4;

  // per-thread staging coordinates
  const int am_st = t >> 1;            // 0..127
  const int ak_st = (t & 1) * 16;      // 0 or 16 -> 16 bf16 per thread
  const __bf16* gA = A + offA + (long long)(m0 + am_st) * sAm + ak_st;
  const int bn_st = t & 63, bk_st = (t >> 6) * 8;   // BPATH 0
  const int bkk = t & 31, bnb = (t >> 5) * 8;       // BPATH 1
  const __bf16* gB0 = B + offB + (long long)(n0 + bn_st) * sBn + bk_st;
  const __bf16* gB1 = B + offB + (long long)bkk * sBk + (n0 + bnb);

  v8f c00 = {}; v8f c01 = {}; v8f c10 = {}; v8f c11 = {};

  const int K32 = K >> 5;

  auto stage = [&](int kt, int buf) {
    const __bf16* a = gA + kt * 32;
    __bf16* la = &As[buf][am_st][ak_st];
    copy16_g2l(a, la);
    copy16_g2l(a + 8, la + 8);
    if (BPATH == 0) {
      copy16_g2l(gB0 + kt * 32, &Bs[buf][bn_st][bk_st]);
    } else {
      const __bf16* gb = gB1 + (long long)kt * 32 * sBk;
      v4u bv = *(const v4u*)gb;                 // 8 contiguous n
      const __bf16* bp = (const __bf16*)&bv;
#pragma unroll
      for (int i = 0; i < 8; ++i) Bs[buf][bnb + i][bkk] = bp[i];  // transpose
    }
  };

  stage(0, 0);
  async_wait_all();
  __syncthreads();

  for (int kt = 0; kt < K32; ++kt) {
    const int cur = kt & 1;
    if (kt + 1 < K32) stage(kt + 1, cur ^ 1);   // overlap next tile

    // ---- fragments (ISA 16-bit layouts) ----
    const __bf16 (*Ab)[40] = As[cur];
    const __bf16 (*Bb)[40] = Bs[cur];
    const int akb = lh * 8;                      // A: K 0..7/16..23 | 8..15/24..31
    const __bf16* pa0 = &Ab[wmBase + l15][akb];
    const __bf16* pa1 = &Ab[wmBase + 16 + l15][akb];
    v16bf a0 = mkfrag(*(const v4u*)pa0, *(const v4u*)(pa0 + 16));
    v16bf a1 = mkfrag(*(const v4u*)pa1, *(const v4u*)(pa1 + 16));
    const int bkb = lh * 8;                      // B: K 0..7/16..23 | 8..15/24..31
    const __bf16* pb0 = &Bb[wnBase + l15][bkb];
    const __bf16* pb1 = &Bb[wnBase + 16 + l15][bkb];
    v16bf b0 = mkfrag(*(const v4u*)pb0, *(const v4u*)(pb0 + 16));
    v16bf b1 = mkfrag(*(const v4u*)pb1, *(const v4u*)(pb1 + 16));

    c00 = __builtin_amdgcn_wmma_f32_16x16x32_bf16(false, a0, false, b0,
                                                  (short)0, c00, false, false);
    c01 = __builtin_amdgcn_wmma_f32_16x16x32_bf16(false, a0, false, b1,
                                                  (short)0, c01, false, false);
    c10 = __builtin_amdgcn_wmma_f32_16x16x32_bf16(false, a1, false, b0,
                                                  (short)0, c10, false, false);
    c11 = __builtin_amdgcn_wmma_f32_16x16x32_bf16(false, a1, false, b1,
                                                  (short)0, c11, false, false);

    if (kt + 1 < K32) {
      async_wait_all();
      __syncthreads();
    }
  }

  // ---- epilogue: C layout VGPR r -> M = r + (lane>=16 ? 8 : 0), N = l15 ----
  const int gmA = m0 + wmBase + lh * 8;   // rows gmA+r (band0), gmA+16+r (band1)
  const int gn  = n0 + wnBase + l15;
  float bi0 = 0.f, bi1 = 0.f;
  if (bias) { bi0 = bias[gn]; bi1 = bias[gn + 16]; }
#pragma unroll
  for (int r = 0; r < 8; ++r) {
    float v00 = c00[r] + bi0, v01 = c01[r] + bi1;
    float v10 = c10[r] + bi0, v11 = c11[r] + bi1;
    if (RELU) {
      v00 = fmaxf(v00, 0.f); v01 = fmaxf(v01, 0.f);
      v10 = fmaxf(v10, 0.f); v11 = fmaxf(v11, 0.f);
    }
    const long long i0 = offC + (long long)(gmA + r) * ldC + gn;
    const long long i1 = offC + (long long)(gmA + 16 + r) * ldC + gn;
    if (Cf)  { Cf[i0] = v00; Cf[i0 + 16] = v01; Cf[i1] = v10; Cf[i1 + 16] = v11; }
    if (Cbf) { Cbf[i0] = f2bf(v00); Cbf[i0 + 16] = f2bf(v01);
               Cbf[i1] = f2bf(v10); Cbf[i1 + 16] = f2bf(v11); }
  }
}

// ---------------------------------------------------------------------------
// softmax over last dim of s[bb,q,0..255]; scale and mask-bias folded in.
// ---------------------------------------------------------------------------
__global__ __launch_bounds__(256)
void softmax_kernel(const float* __restrict__ s,
                    const unsigned char* __restrict__ mask,
                    __bf16* __restrict__ p) {
  __shared__ float red[256];
  const int q = blockIdx.x, bb = blockIdx.y;
  const int b = bb >> 3;
  const int j = threadIdx.x;
  const long long base = ((long long)bb * 256 + q) * 256;
  const float v = s[base + j] * 0.125f + (mask[b * 256 + j] ? 0.f : -1e9f);
  red[j] = v; __syncthreads();
  for (int o = 128; o > 0; o >>= 1) {
    if (j < o) red[j] = fmaxf(red[j], red[j + o]);
    __syncthreads();
  }
  const float mx = red[0]; __syncthreads();
  const float e = __expf(v - mx);
  red[j] = e; __syncthreads();
  for (int o = 128; o > 0; o >>= 1) {
    if (j < o) red[j] += red[j + o];
    __syncthreads();
  }
  p[base + j] = f2bf(e / red[0]);
}

// ---------------------------------------------------------------------------
// x = LayerNorm(x + r) * g + b ; writes f32 master + bf16 copy. Row = 512.
// ---------------------------------------------------------------------------
__global__ __launch_bounds__(256)
void add_ln_kernel(float* __restrict__ x, const float* __restrict__ r,
                   const float* __restrict__ g, const float* __restrict__ bta,
                   __bf16* __restrict__ xbf) {
  __shared__ float s1[256], s2[256];
  const int row = blockIdx.x, tid = threadIdx.x;
  const long long o = (long long)row * 512;
  const float v0 = x[o + tid]       + r[o + tid];
  const float v1 = x[o + tid + 256] + r[o + tid + 256];
  s1[tid] = v0 + v1; s2[tid] = v0 * v0 + v1 * v1;
  __syncthreads();
  for (int of = 128; of > 0; of >>= 1) {
    if (tid < of) { s1[tid] += s1[tid + of]; s2[tid] += s2[tid + of]; }
    __syncthreads();
  }
  const float mean = s1[0] * (1.f / 512.f);
  const float var  = s2[0] * (1.f / 512.f) - mean * mean;
  const float rs   = rsqrtf(var + 1e-5f);
  const float y0 = (v0 - mean) * rs * g[tid]       + bta[tid];
  const float y1 = (v1 - mean) * rs * g[tid + 256] + bta[tid + 256];
  x[o + tid] = y0;           x[o + tid + 256] = y1;
  xbf[o + tid] = f2bf(y0);   xbf[o + tid + 256] = f2bf(y1);
}

// ---------------------------------------------------------------------------
// Host-side orchestration
// ---------------------------------------------------------------------------
extern "C" void kernel_launch(void* const* d_in, const int* in_sizes, int n_in,
                              void* d_out, int out_size, void* d_ws, size_t ws_size,
                              hipStream_t stream) {
  const float*         attn = (const float*)d_in[0];
  const unsigned char* mask = (const unsigned char*)d_in[1];
  const int*           tgt  = (const int*)d_in[2];
  const float*         emb  = (const float*)d_in[3];
  const float* Wq   = (const float*)d_in[4];
  const float* Wk   = (const float*)d_in[5];
  const float* Wv   = (const float*)d_in[6];
  const float* Wo   = (const float*)d_in[7];
  const float* ln1g = (const float*)d_in[8];
  const float* ln1b = (const float*)d_in[9];
  const float* W1   = (const float*)d_in[10];
  const float* b1   = (const float*)d_in[11];
  const float* W2   = (const float*)d_in[12];
  const float* b2   = (const float*)d_in[13];
  const float* ln2g = (const float*)d_in[14];
  const float* ln2b = (const float*)d_in[15];
  const float* proj = (const float*)d_in[16];
  float* out = (float*)d_out;

  char* wp = (char*)d_ws;
  auto alloc = [&](size_t n) { char* r = wp; wp += (n + 255) & ~(size_t)255; return r; };
  float*  xf   = (float*)alloc(2048ll * 512 * 4);
  float*  tmp  = (float*)alloc(2048ll * 512 * 4);
  float*  sbuf = (float*)alloc(64ll * 256 * 256 * 4);
  __bf16* xb   = (__bf16*)alloc(2048ll * 512 * 2);
  __bf16* qb   = (__bf16*)alloc(2048ll * 512 * 2);
  __bf16* kb   = (__bf16*)alloc(2048ll * 512 * 2);
  __bf16* vb   = (__bf16*)alloc(2048ll * 512 * 2);
  __bf16* ob   = (__bf16*)alloc(2048ll * 512 * 2);
  __bf16* f1b  = (__bf16*)alloc(2048ll * 2048 * 2);
  __bf16* pb   = (__bf16*)alloc(64ll * 256 * 256 * 2);
  __bf16* wqb  = (__bf16*)alloc(4ll * 512 * 512 * 2);
  __bf16* wkb  = (__bf16*)alloc(4ll * 512 * 512 * 2);
  __bf16* wvb  = (__bf16*)alloc(4ll * 512 * 512 * 2);
  __bf16* wob  = (__bf16*)alloc(4ll * 512 * 512 * 2);
  __bf16* w1b  = (__bf16*)alloc(4ll * 512 * 2048 * 2);
  __bf16* w2b  = (__bf16*)alloc(4ll * 2048 * 512 * 2);
  __bf16* pjb  = (__bf16*)alloc(16000ll * 512 * 2);

  auto cvt = [&](const float* s, __bf16* d, long long n) {
    int grid = (int)((n + 1023) / 1024); if (grid > 4096) grid = 4096;
    cvt_kernel<<<dim3(grid), dim3(256), 0, stream>>>(s, d, n);
  };
  cvt(Wq, wqb, 4ll * 512 * 512);
  cvt(Wk, wkb, 4ll * 512 * 512);
  cvt(Wv, wvb, 4ll * 512 * 512);
  cvt(Wo, wob, 4ll * 512 * 512);
  cvt(W1, w1b, 4ll * 512 * 2048);
  cvt(W2, w2b, 4ll * 2048 * 512);
  cvt(proj, pjb, 16000ll * 512);

  embed_kernel<<<dim3(2048), dim3(256), 0, stream>>>(attn, tgt, emb, xf, xb);

  // bpath: 0 => sBk==1 (k contiguous), 1 => sBn==1 (n contiguous)
  auto gemm = [&](int bpath, bool relu,
                  const __bf16* A, const __bf16* B, const float* bias,
                  float* Cf, __bf16* Cbf, int M, int N, int K,
                  int sAm, long long bsAo, long long bsAi,
                  int sBk, int sBn, long long bsBo, long long bsBi,
                  int ldC, long long bsCo, long long bsCi,
                  int nb, int innerB) {
    dim3 g(N / 64, M / 128, nb);
    if (bpath == 0)
      gemm_bf16_kernel<0, false><<<g, dim3(256), 0, stream>>>(
          A, B, bias, Cf, Cbf, M, N, K, sAm, bsAo, bsAi,
          sBk, sBn, bsBo, bsBi, ldC, bsCo, bsCi, innerB);
    else if (relu)
      gemm_bf16_kernel<1, true><<<g, dim3(256), 0, stream>>>(
          A, B, bias, Cf, Cbf, M, N, K, sAm, bsAo, bsAi,
          sBk, sBn, bsBo, bsBi, ldC, bsCo, bsCi, innerB);
    else
      gemm_bf16_kernel<1, false><<<g, dim3(256), 0, stream>>>(
          A, B, bias, Cf, Cbf, M, N, K, sAm, bsAo, bsAi,
          sBk, sBn, bsBo, bsBi, ldC, bsCo, bsCi, innerB);
  };

  for (int i = 0; i < 4; ++i) {
    const __bf16* wq = wqb + (long long)i * 512 * 512;
    const __bf16* wk = wkb + (long long)i * 512 * 512;
    const __bf16* wv = wvb + (long long)i * 512 * 512;
    const __bf16* wo = wob + (long long)i * 512 * 512;
    // q,k,v = x @ W   (M=2048, N=512, K=512); W is KxN row-major (sBn==1)
    gemm(1, false, xb, wq, nullptr, nullptr, qb, 2048, 512, 512,
         512, 0, 0, 512, 1, 0, 0, 512, 0, 0, 1, 1);
    gemm(1, false, xb, wk, nullptr, nullptr, kb, 2048, 512, 512,
         512, 0, 0, 512, 1, 0, 0, 512, 0, 0, 1, 1);
    gemm(1, false, xb, wv, nullptr, nullptr, vb, 2048, 512, 512,
         512, 0, 0, 512, 1, 0, 0, 512, 0, 0, 1, 1);
    // scores = q @ k^T per (b,h): 64 batches, M=N=256, K=64 (B: sBk==1)
    gemm(0, false, qb, kb, nullptr, sbuf, nullptr, 256, 256, 64,
         512, 131072, 64, 1, 512, 131072, 64, 256, 524288, 65536, 64, 8);
    softmax_kernel<<<dim3(256, 64), dim3(256), 0, stream>>>(sbuf, mask, pb);
    // o = probs @ v per (b,h): M=256, N=64, K=256 (B: sBn==1)
    gemm(1, false, pb, vb, nullptr, nullptr, ob, 256, 64, 256,
         256, 524288, 65536, 512, 1, 131072, 64, 512, 131072, 64, 64, 8);
    // attn_out = o @ Wo
    gemm(1, false, ob, wo, nullptr, tmp, nullptr, 2048, 512, 512,
         512, 0, 0, 512, 1, 0, 0, 512, 0, 0, 1, 1);
    add_ln_kernel<<<dim3(2048), dim3(256), 0, stream>>>(
        xf, tmp, ln1g + i * 512, ln1b + i * 512, xb);
    // FFN
    gemm(1, true, xb, w1b + (long long)i * 512 * 2048, b1 + i * 2048,
         nullptr, f1b, 2048, 2048, 512,
         512, 0, 0, 2048, 1, 0, 0, 2048, 0, 0, 1, 1);
    gemm(1, false, f1b, w2b + (long long)i * 2048 * 512, b2 + i * 512,
         tmp, nullptr, 2048, 512, 2048,
         2048, 0, 0, 512, 1, 0, 0, 512, 0, 0, 1, 1);
    add_ln_kernel<<<dim3(2048), dim3(256), 0, stream>>>(
        xf, tmp, ln2g + i * 512, ln2b + i * 512, xb);
  }

  // logits = x @ proj^T : B[k][n] = proj[n*512 + k] (sBk==1)
  gemm(0, false, xb, pjb, nullptr, out, nullptr, 2048, 16000, 512,
       512, 0, 0, 1, 512, 0, 0, 16000, 0, 0, 1, 1);
}
